// TinyVitAttention_63814624084644
// MI455X (gfx1250) — matrix operand order repaired and checked by reading, not compiled, not run on hardware
//
#include <hip/hip_runtime.h>
#include <hip/hip_bf16.h>

typedef __attribute__((ext_vector_type(16))) _Float16 v16h;
typedef __attribute__((ext_vector_type(8)))  float    v8f;

#define NTOK   196
#define NHEAD  18
#define KD     32
#define VD     128
#define HSL    192            // per-head slice inside a qkv row (32q+32k+128v)
#define QKVN   3456           // qkv row width
#define DIMC   576
#define DHC    2304
#define MROWS  25088          // B*N = 128*196
#define ATTN_SCALE 0.17677669529663687f

union Frag {
    v16h  v;
    float4 q[2];
};

// ---------------------------------------------------------------- f32 -> f16
__global__ void cvt_f16_kernel(const float* __restrict__ in, _Float16* __restrict__ o, int n)
{
    int i = blockIdx.x * 256 + threadIdx.x;
    if (i < n) o[i] = (_Float16)in[i];
}

// ---------------------------------------------------------------- LayerNorm (one wave per row)
__global__ __launch_bounds__(256) void ln_kernel(const float* __restrict__ x,
                                                 const float* __restrict__ w,
                                                 const float* __restrict__ bvec,
                                                 _Float16* __restrict__ y)
{
    int row  = blockIdx.x * 8 + (threadIdx.x >> 5);
    int lane = threadIdx.x & 31;
    const float* xr = x + (size_t)row * DIMC;
    float vals[18];
    float s = 0.f, s2 = 0.f;
#pragma unroll
    for (int j = 0; j < 18; ++j) {
        float v = xr[lane + j * 32];
        vals[j] = v; s += v; s2 += v * v;
    }
#pragma unroll
    for (int off = 16; off > 0; off >>= 1) {
        s  += __shfl_xor(s,  off, 32);
        s2 += __shfl_xor(s2, off, 32);
    }
    float mu  = s * (1.0f / DIMC);
    float inv = rsqrtf(s2 * (1.0f / DIMC) - mu * mu + 1e-5f);
    _Float16* yr = y + (size_t)row * DIMC;
#pragma unroll
    for (int j = 0; j < 18; ++j) {
        int c = lane + j * 32;
        yr[c] = (_Float16)((vals[j] - mu) * inv * w[c] + bvec[c]);
    }
}

// ---------------------------------------------------------------- bias table expand: bias_h[h][n][m]
__global__ void bias_expand_kernel(const float* __restrict__ ab,
                                   const int* __restrict__ idxs,
                                   float* __restrict__ outb)
{
    int i = blockIdx.x * 256 + threadIdx.x;
    if (i < NTOK * NTOK) {
        int id = idxs[i];
#pragma unroll
        for (int h = 0; h < NHEAD; ++h)
            outb[h * NTOK * NTOK + i] = ab[h * NTOK + id];
    }
}

// ---------------------------------------------------------------- generic  out = A * W^T + bias
// A: (M x K) f16 row-major, W: (N x K) f16 row-major. Block = 8 waves -> 128x64 tile.
// K must be a multiple of 64 (576 and 2304 both are). Software-pipelined 2 k-steps deep.
__global__ __launch_bounds__(256) void gemm_wmma_kernel(const _Float16* __restrict__ A,
                                                        const _Float16* __restrict__ W,
                                                        const float* __restrict__ bias,
                                                        void* __restrict__ out,
                                                        int N, int K, int out_f16)
{
    const int lane  = threadIdx.x & 31;
    const int wave  = threadIdx.x >> 5;
    const int halfk = lane >> 4;
    const int l16   = lane & 15;
    const int m0    = blockIdx.y * 128 + wave * 16;
    const int n0    = blockIdx.x * 64;

    const _Float16* arow = A + (size_t)(m0 + l16) * K;
    const _Float16* wb0  = W + (size_t)(n0 +  0 + l16) * K + 16 * halfk;
    const _Float16* wb1  = W + (size_t)(n0 + 16 + l16) * K + 16 * halfk;
    const _Float16* wb2  = W + (size_t)(n0 + 32 + l16) * K + 16 * halfk;
    const _Float16* wb3  = W + (size_t)(n0 + 48 + l16) * K + 16 * halfk;

    v8f acc[4];
#pragma unroll
    for (int t = 0; t < 4; ++t)
#pragma unroll
        for (int i = 0; i < 8; ++i) acc[t][i] = 0.0f;

    Frag a0, a1, b0[4], b1[4];

    auto load_step = [&](Frag& a, Frag* bf, int k0) {
        a.q[0] = *(const float4*)(arow + k0 + 16 * halfk);
        a.q[1] = *(const float4*)(arow + k0 + 16 * halfk + 8);
        bf[0].q[0] = *(const float4*)(wb0 + k0);
        bf[0].q[1] = *(const float4*)(wb0 + k0 + 8);
        bf[1].q[0] = *(const float4*)(wb1 + k0);
        bf[1].q[1] = *(const float4*)(wb1 + k0 + 8);
        bf[2].q[0] = *(const float4*)(wb2 + k0);
        bf[2].q[1] = *(const float4*)(wb2 + k0 + 8);
        bf[3].q[0] = *(const float4*)(wb3 + k0);
        bf[3].q[1] = *(const float4*)(wb3 + k0 + 8);
    };

    load_step(a0, b0, 0);
    for (int k0 = 0; k0 < K; k0 += 64) {
        load_step(a1, b1, k0 + 32);            // prefetch step k0+32 while computing k0
#pragma unroll
        for (int t = 0; t < 4; ++t)
            acc[t] = __builtin_amdgcn_wmma_f32_16x16x32_f16(false, a0.v, false, b0[t].v,
                                                            (short)0, acc[t], false, false);
        if (k0 + 64 < K) load_step(a0, b0, k0 + 64);   // prefetch step k0+64
#pragma unroll
        for (int t = 0; t < 4; ++t)
            acc[t] = __builtin_amdgcn_wmma_f32_16x16x32_f16(false, a1.v, false, b1[t].v,
                                                            (short)0, acc[t], false, false);
    }

#pragma unroll
    for (int t = 0; t < 4; ++t) {
        const int col = n0 + t * 16 + l16;
        const float bv = bias[col];
#pragma unroll
        for (int r = 0; r < 8; ++r) {
            const int row = m0 + r + 8 * halfk;
            const float v = acc[t][r] + bv;
            if (out_f16) ((_Float16*)out)[(size_t)row * N + col] = (_Float16)v;
            else         ((float*)out)[(size_t)row * N + col]    = v;
        }
    }
}

// ---------------------------------------------------------------- fused attention per (b,h)
// LDS: vt[128][224] f16 (V transposed)  +  pbuf[8 waves][16][224] f16 (P scratch)
#define ATTN_SMEM ((128 * 224 + 8 * 16 * 224) * 2)

__global__ __launch_bounds__(256) void attn_kernel(const _Float16* __restrict__ qkv,
                                                   const float* __restrict__ bias_h,
                                                   _Float16* __restrict__ ctx)
{
    extern __shared__ _Float16 smem[];
    _Float16* vt   = smem;                 // [128][224]
    _Float16* pbuf = smem + 128 * 224;     // [8][16][224]

    const int h    = blockIdx.x;
    const int b    = blockIdx.y;
    const int tid  = threadIdx.x;
    const int lane = tid & 31;
    const int wave = tid >> 5;
    const int halfk = lane >> 4;
    const int l16   = lane & 15;

    const _Float16* base = qkv + (size_t)b * NTOK * QKVN + h * HSL;

    // zero V-token padding (196..223), then transpose V into LDS
    for (int i = tid; i < 128 * 28; i += 256) {
        int d = i / 28, tok = 196 + i % 28;
        vt[d * 224 + tok] = (_Float16)0.0f;
    }
    for (int t = tid; t < 196 * 128; t += 256) {
        int tok = t >> 7, d = t & 127;
        vt[d * 224 + tok] = base[(size_t)tok * QKVN + 64 + d];
    }
    // zero P column padding (208..223)
    for (int i = tid; i < 8 * 16 * 16; i += 256) {
        int rr = i >> 4, c = 208 + (i & 15);
        pbuf[rr * 224 + c] = (_Float16)0.0f;
    }
    __syncthreads();

    _Float16* myp   = pbuf + wave * 16 * 224;
    const float* bh = bias_h + (size_t)h * NTOK * NTOK;
    _Float16* ctxb  = ctx + (size_t)b * NTOK * DHC + h * VD;

    for (int strip = wave; strip < 13; strip += 8) {
        // ---------------- S = Q K^T  (one 16x16x32 WMMA per key tile, 2-deep pipelined)
        const int mrow   = strip * 16 + l16;
        const int mclamp = mrow < NTOK ? mrow : NTOK - 1;
        Frag qa;
        const _Float16* qrow = base + (size_t)mclamp * QKVN;
        qa.q[0] = *(const float4*)(qrow + 16 * halfk);
        qa.q[1] = *(const float4*)(qrow + 16 * halfk + 8);

        Frag kb[2];
        auto load_k = [&](Frag& f, int j) {
            int col    = j * 16 + l16;
            int cclamp = col < NTOK ? col : NTOK - 1;
            const _Float16* krow = base + (size_t)cclamp * QKVN + 32 + 16 * halfk;
            f.q[0] = *(const float4*)(krow);
            f.q[1] = *(const float4*)(krow + 8);
        };

        v8f s[13];
        load_k(kb[0], 0);
#pragma unroll
        for (int j = 0; j < 13; ++j) {
            if (j < 12) load_k(kb[(j + 1) & 1], j + 1);
            v8f z;
#pragma unroll
            for (int i = 0; i < 8; ++i) z[i] = 0.0f;
            s[j] = __builtin_amdgcn_wmma_f32_16x16x32_f16(false, qa.v, false, kb[j & 1].v,
                                                          (short)0, z, false, false);
        }

        // ---------------- scale + bias + mask, row max
        float rmax[8];
#pragma unroll
        for (int r = 0; r < 8; ++r) rmax[r] = -3.0e38f;
#pragma unroll
        for (int j = 0; j < 13; ++j) {
            int col  = j * 16 + l16;
            bool cok = col < NTOK;
#pragma unroll
            for (int r = 0; r < 8; ++r) {
                int rowt = strip * 16 + r + 8 * halfk;
                float bv = (cok && rowt < NTOK) ? bh[rowt * NTOK + col] : 0.0f;
                float v  = cok ? (s[j][r] * ATTN_SCALE + bv) : -1.0e30f;
                s[j][r]  = v;
                rmax[r]  = fmaxf(rmax[r], v);
            }
        }
#pragma unroll
        for (int r = 0; r < 8; ++r)
#pragma unroll
            for (int off = 8; off > 0; off >>= 1)
                rmax[r] = fmaxf(rmax[r], __shfl_xor(rmax[r], off, 16));

        // ---------------- exp + row sum
        float rsum[8];
#pragma unroll
        for (int r = 0; r < 8; ++r) rsum[r] = 0.0f;
#pragma unroll
        for (int j = 0; j < 13; ++j)
#pragma unroll
            for (int r = 0; r < 8; ++r) {
                float e = __expf(s[j][r] - rmax[r]);
                s[j][r] = e;
                rsum[r] += e;
            }
#pragma unroll
        for (int r = 0; r < 8; ++r)
#pragma unroll
            for (int off = 8; off > 0; off >>= 1)
                rsum[r] += __shfl_xor(rsum[r], off, 16);

        // ---------------- P -> LDS (unnormalized; per-wave region, DS in-order within wave)
#pragma unroll
        for (int j = 0; j < 13; ++j)
#pragma unroll
            for (int r = 0; r < 8; ++r)
                myp[(r + 8 * halfk) * 224 + j * 16 + l16] = (_Float16)s[j][r];

        // ---------------- O = P V  (2-deep pipelined on both LDS operand streams)
        v8f oacc[8];
#pragma unroll
        for (int t = 0; t < 8; ++t)
#pragma unroll
            for (int i = 0; i < 8; ++i) oacc[t][i] = 0.0f;

        Frag pa[2], vb[2];
        auto load_p = [&](Frag& f, int kk) {
            const _Float16* prow = myp + l16 * 224 + kk * 32;
            f.q[0] = *(const float4*)(prow + 16 * halfk);
            f.q[1] = *(const float4*)(prow + 16 * halfk + 8);
        };
        auto load_v = [&](Frag& f, int kk, int t) {
            const _Float16* vrow = vt + (t * 16 + l16) * 224 + kk * 32 + 16 * halfk;
            f.q[0] = *(const float4*)(vrow);
            f.q[1] = *(const float4*)(vrow + 8);
        };

        load_p(pa[0], 0);
        load_v(vb[0], 0, 0);
#pragma unroll
        for (int kk = 0; kk < 7; ++kk) {
            if (kk < 6) load_p(pa[(kk + 1) & 1], kk + 1);
#pragma unroll
            for (int t = 0; t < 8; ++t) {
                if (t < 7)          load_v(vb[(t + 1) & 1], kk, t + 1);
                else if (kk < 6)    load_v(vb[0], kk + 1, 0);
                oacc[t] = __builtin_amdgcn_wmma_f32_16x16x32_f16(false, pa[kk & 1].v, false,
                                                                 vb[t & 1].v, (short)0,
                                                                 oacc[t], false, false);
            }
        }

        // ---------------- normalize + store ctx (f16)
#pragma unroll
        for (int t = 0; t < 8; ++t) {
            int d = t * 16 + l16;
#pragma unroll
            for (int r = 0; r < 8; ++r) {
                int rowt = strip * 16 + r + 8 * halfk;
                if (rowt < NTOK)
                    ctxb[(size_t)rowt * DHC + d] = (_Float16)(oacc[t][r] / rsum[r]);
            }
        }
    }
}

// ---------------------------------------------------------------- launch
extern "C" void kernel_launch(void* const* d_in, const int* in_sizes, int n_in,
                              void* d_out, int out_size, void* d_ws, size_t ws_size,
                              hipStream_t stream)
{
    const float* hidden      = (const float*)d_in[0];
    const float* ln_w        = (const float*)d_in[1];
    const float* ln_b        = (const float*)d_in[2];
    const float* qkv_w       = (const float*)d_in[3];
    const float* qkv_b       = (const float*)d_in[4];
    const float* proj_w      = (const float*)d_in[5];
    const float* proj_b      = (const float*)d_in[6];
    const float* attn_biases = (const float*)d_in[7];
    const int*   bias_idxs   = (const int*)d_in[8];
    float* out = (float*)d_out;

    char* ws = (char*)d_ws;
    size_t off = 0;
    auto take = [&](size_t bytes) {
        char* p = ws + off;
        off = (off + bytes + 255) & ~(size_t)255;
        return p;
    };
    _Float16* qkv_w_h  = (_Float16*)take((size_t)QKVN * DIMC * 2);
    _Float16* proj_w_h = (_Float16*)take((size_t)DIMC * DHC * 2);
    _Float16* x_h      = (_Float16*)take((size_t)MROWS * DIMC * 2);
    float*    bias_h   = (float*)   take((size_t)NHEAD * NTOK * NTOK * 4);
    _Float16* qkv_h    = (_Float16*)take((size_t)MROWS * QKVN * 2);
    _Float16* ctx_h    = (_Float16*)take((size_t)MROWS * DHC * 2);

    // 1. weight conversions to f16
    {
        int n1 = QKVN * DIMC;
        cvt_f16_kernel<<<(n1 + 255) / 256, 256, 0, stream>>>(qkv_w, qkv_w_h, n1);
        int n2 = DIMC * DHC;
        cvt_f16_kernel<<<(n2 + 255) / 256, 256, 0, stream>>>(proj_w, proj_w_h, n2);
    }
    // 2. LayerNorm -> f16 activations
    ln_kernel<<<MROWS / 8, 256, 0, stream>>>(hidden, ln_w, ln_b, x_h);
    // 3. expand attention bias table
    bias_expand_kernel<<<(NTOK * NTOK + 255) / 256, 256, 0, stream>>>(attn_biases, bias_idxs, bias_h);
    // 4. QKV GEMM (25088 x 3456 x 576), f16 output
    gemm_wmma_kernel<<<dim3(QKVN / 64, MROWS / 128), 256, 0, stream>>>(
        x_h, qkv_w_h, qkv_b, (void*)qkv_h, QKVN, DIMC, 1);
    // 5. fused attention per (b, h)
    (void)hipFuncSetAttribute(reinterpret_cast<const void*>(attn_kernel),
                              hipFuncAttributeMaxDynamicSharedMemorySize, ATTN_SMEM);
    attn_kernel<<<dim3(NHEAD, 128), 256, ATTN_SMEM, stream>>>(qkv_h, bias_h, ctx_h);
    // 6. projection GEMM (25088 x 576 x 2304), f32 output + bias
    gemm_wmma_kernel<<<dim3(DIMC / 64, MROWS / 128), 256, 0, stream>>>(
        ctx_h, proj_w_h, proj_b, (void*)out, DIMC, DHC, 0);
}
